// RecursiveDecoder_26577257628371
// MI455X (gfx1250) — matrix operand order repaired and checked by reading, not compiled, not run on hardware
//
#include <hip/hip_runtime.h>

// ---------------------------------------------------------------------------
// Problem constants (match reference)
// ---------------------------------------------------------------------------
constexpr int kM   = 100;   // max_child_num
constexpr int kMP  = 112;   // padded to multiple of 16
constexpr int kH   = 512;   // hidden
constexpr int kE   = 512;   // edge hidden
constexpr int kT   = 4;     // edge types
constexpr int kIT  = 2;     // iterations
constexpr int kSEM = 57;
constexpr int kSEMP = 64;   // padded sem cols

typedef __attribute__((ext_vector_type(16))) _Float16 v16h;
typedef __attribute__((ext_vector_type(8)))  _Float16 v8h;
typedef __attribute__((ext_vector_type(8)))  float    v8f;

// Build a v16h WMMA fragment from two contiguous 16-byte chunks at p and
// p+gap (gap in halves). A and B fragments both use gap=16.
__device__ inline v16h load_frag(const _Float16* __restrict__ p, int gap) {
  v8h lo = *(const v8h*)(p);
  v8h hi = *(const v8h*)(p + gap);
  return __builtin_shufflevector(lo, hi, 0,1,2,3,4,5,6,7,8,9,10,11,12,13,14,15);
}

// ---------------------------------------------------------------------------
// Wave-level 16x16 WMMA tile: C(16x16,f32) = A(16xK,f16 row-major) *
// Bt(K x 16 via pre-transposed f16, column n contiguous in K).
// Fragment layouts per CDNA5 ISA 7.12.2 (wave32):
//   A 16-bit 16x32: lanes 0-15 row=lane, elems = K {0..7,16..23}; lanes 16-31
//                   row=lane-16, elems = K {8..15,24..31}
//   B 32x16:        lanes 0-15 col=lane, K {0..7,16..23}; lanes 16-31 col, K {8..15,24..31}
//   C/D f32:        vgpr v, lanes 0-15 -> row v, lanes 16-31 -> row v+8
// ---------------------------------------------------------------------------
__device__ inline v8f wmma_tile(const _Float16* __restrict__ A, int lda,
                                const _Float16* __restrict__ Bt, int ldb,
                                int K, int lane) {
  const int hs = (lane >> 4) & 1;   // half-select
  const int lr = lane & 15;
  const _Float16* __restrict__ arow = A  + (size_t)lr * lda + hs * 8;
  const _Float16* __restrict__ bcol = Bt + (size_t)lr * ldb + hs * 8;
  v8f c = {};
  for (int k0 = 0; k0 < K; k0 += 32) {
    v16h a = load_frag(arow + k0, 16);
    v16h b = load_frag(bcol + k0, 16);
    c = __builtin_amdgcn_wmma_f32_16x16x32_f16(false, a, false, b, (short)0, c,
                                               false, false);
  }
  return c;
}

// ---------------------------------------------------------------------------
// Generic GEMM: C[Mtiles*16, Ntiles*16] = A(f16) * Bt(f16,T) (+bias)(+relu)
// (used only for small 112-row GEMMs; operand traffic is trivial there)
// ---------------------------------------------------------------------------
__global__ void gemm_kernel(const _Float16* __restrict__ A, int lda,
                            const _Float16* __restrict__ Bt, int ldb,
                            float* __restrict__ C, int ldc,
                            int Mtiles, int Ntiles, int K,
                            const float* __restrict__ bias, int nbias, int relu) {
  int wave = (int)((blockIdx.x * blockDim.x + threadIdx.x) >> 5);
  int lane = threadIdx.x & 31;
  if (wave >= Mtiles * Ntiles) return;          // whole-wave uniform exit
  int mt = wave / Ntiles, nt = wave % Ntiles;
  int hs = (lane >> 4) & 1, lr = lane & 15;
  v8f c = wmma_tile(A + (size_t)(mt * 16) * lda, lda,
                    Bt + (size_t)(nt * 16) * ldb, ldb, K, lane);
  int col = nt * 16 + lr;
  float bv = (bias != nullptr && col < nbias) ? bias[col] : 0.f;
#pragma unroll
  for (int v = 0; v < 8; ++v) {
    int row = mt * 16 + hs * 8 + v;
    float x = c[v] + bv;
    if (relu) x = fmaxf(x, 0.f);
    C[(size_t)row * ldc + col] = x;
  }
}

// ---------------------------------------------------------------------------
// child0 = relu(parent @ W_parent + b_parent)   (GEMV, streams 105MB)
// ---------------------------------------------------------------------------
__global__ void parent_kernel(const float* __restrict__ pf,
                              const float* __restrict__ Wp,
                              const float* __restrict__ bp,
                              float* __restrict__ child0) {
  int q = blockIdx.x * blockDim.x + threadIdx.x;
  if (q >= (kM * kH) / 4) return;
  int c0 = q * 4;
  float4 acc = *(const float4*)(bp + c0);
  const float* w = Wp + c0;
  for (int k = 0; k < kH; ++k) {
    float p = pf[k];
    float4 wv = *(const float4*)(w + (size_t)k * (kM * kH));
    acc.x = fmaf(p, wv.x, acc.x);
    acc.y = fmaf(p, wv.y, acc.y);
    acc.z = fmaf(p, wv.z, acc.z);
    acc.w = fmaf(p, wv.w, acc.w);
  }
  float4 r;
  r.x = fmaxf(acc.x, 0.f); r.y = fmaxf(acc.y, 0.f);
  r.z = fmaxf(acc.z, 0.f); r.w = fmaxf(acc.w, 0.f);
  *(float4*)(child0 + c0) = r;
}

// ---------------------------------------------------------------------------
// exists_logits + node-ok mask (one wave per row); also zero any_edge
// ---------------------------------------------------------------------------
__global__ void exists_kernel(const float* __restrict__ child0,
                              const float* __restrict__ Wex,
                              const float* __restrict__ bex,
                              float* __restrict__ exists_out,
                              unsigned char* __restrict__ ok,
                              int* __restrict__ any_edge) {
  int wv = (int)((blockIdx.x * blockDim.x + threadIdx.x) >> 5);
  int lane = threadIdx.x & 31;
  if (blockIdx.x == 0 && threadIdx.x == 0) *any_edge = 0;
  if (wv >= kMP) return;
  if (wv >= kM) { if (lane == 0) ok[wv] = 0; return; }
  float s = 0.f;
  for (int e = lane; e < kH; e += 32) s += child0[(size_t)wv * kH + e] * Wex[e];
  for (int o = 16; o; o >>= 1) s += __shfl_xor(s, o, 32);
  if (lane == 0) {
    float v = s + bex[0];
    exists_out[wv] = v;
    ok[wv] = (v > 0.f) ? 1 : 0;
  }
}

// ---------------------------------------------------------------------------
// edge latents (f16) + edge-exists logits + any_edge.  block = (i,j), 256 thr
// EL16 layout: [kM][kMP][kE] (padded j rows zeroed)
// ---------------------------------------------------------------------------
__global__ void edge_kernel(const float* __restrict__ Ael,   // bias folded
                            const float* __restrict__ Bel,
                            const float* __restrict__ Wee,   // [4][512]
                            const float* __restrict__ bee,
                            _Float16* __restrict__ EL16,
                            float* __restrict__ logits,      // ws copy
                            float* __restrict__ logits_out,  // d_out region
                            const unsigned char* __restrict__ ok,
                            int* __restrict__ any_edge) {
  const int i = blockIdx.x;
  const int j = blockIdx.y;
  const int tid = threadIdx.x;
  _Float16* dst = EL16 + ((size_t)i * kMP + j) * kE;
  if (j >= kM) {
    for (int e = tid; e < kE; e += 256) dst[e] = (_Float16)0.f;
    return;
  }
  const int e = tid * 2;
  float v0 = fmaxf(Ael[(size_t)i * kE + e]     + Bel[(size_t)j * kE + e],     0.f);
  float v1 = fmaxf(Ael[(size_t)i * kE + e + 1] + Bel[(size_t)j * kE + e + 1], 0.f);
  dst[e]     = (_Float16)v0;
  dst[e + 1] = (_Float16)v1;
  float pt[kT];
#pragma unroll
  for (int t = 0; t < kT; ++t)
    pt[t] = v0 * Wee[t * kE + e] + v1 * Wee[t * kE + e + 1];
#pragma unroll
  for (int t = 0; t < kT; ++t)
    for (int o = 16; o; o >>= 1) pt[t] += __shfl_xor(pt[t], o, 32);
  __shared__ float red[8][kT];
  int wv = tid >> 5, lane = tid & 31;
  if (lane == 0) {
#pragma unroll
    for (int t = 0; t < kT; ++t) red[wv][t] = pt[t];
  }
  __syncthreads();
  if (tid < kT) {
    float s = 0.f;
    for (int w = 0; w < 8; ++w) s += red[w][tid];
    float lg = s + bee[tid];
    size_t o4 = ((size_t)i * kM + j) * kT + tid;
    logits[o4] = lg;
    logits_out[o4] = lg;
    if (lg > 0.f && ok[i] && ok[j]) atomicOr(any_edge, 1);
  }
}

// ---------------------------------------------------------------------------
// Fused message passing for one iteration (k-outer, persistent accumulators).
//   new_child[i,h] = sum_{j,t} relu(P[i,h]+Q[j,h]+(EL[i,j]·We)[h]+lg*Wt[t,h])
//                    masked;  child_next = any_edge ? new : prev
// grid (kM, 2), block 256 = 8 waves. Each wave owns TWO 16-col tiles and
// keeps all 7 j-chunk accumulators live (v8f c[2][7]) so every B fragment is
// loaded exactly once per wave and each A fragment feeds 2 WMMAs.
// ---------------------------------------------------------------------------
__global__ void msg_kernel(const _Float16* __restrict__ EL16,
                           const _Float16* __restrict__ We_t,  // [512][512] f16
                           const float* __restrict__ P,        // bias folded
                           const float* __restrict__ Q,
                           const float* __restrict__ Wt,       // [4][512] f32
                           const float* __restrict__ logits,
                           const unsigned char* __restrict__ ok,
                           const int* __restrict__ any_edge,
                           const float* __restrict__ child_prev,
                           float* __restrict__ child_next) {
  const int i = blockIdx.x;
  const int wv = threadIdx.x >> 5;
  const int lane = threadIdx.x & 31;
  const int hs = (lane >> 4) & 1;
  const int lr = lane & 15;
  const int tbase = blockIdx.y * 256 + wv * 32;  // base of the 2-tile group
  const int col0 = tbase + lr;
  const int col1 = col0 + 16;
  float acc0 = 0.f, acc1 = 0.f;
  if (ok[i]) {
    v8f c[2][7];
#pragma unroll
    for (int n = 0; n < 2; ++n)
#pragma unroll
      for (int jb = 0; jb < 7; ++jb) c[n][jb] = {};
    const _Float16* __restrict__ Abase =
        EL16 + (size_t)i * kMP * kE + (size_t)lr * kE + hs * 8;
    const _Float16* __restrict__ b0p =
        We_t + (size_t)(tbase + lr) * kE + hs * 8;
    const _Float16* __restrict__ b1p = b0p + (size_t)16 * kE;
    for (int k0 = 0; k0 < kE; k0 += 32) {
      v16h b0 = load_frag(b0p + k0, 16);
      v16h b1 = load_frag(b1p + k0, 16);
#pragma unroll
      for (int jb = 0; jb < 7; ++jb) {
        v16h a = load_frag(Abase + (size_t)jb * 16 * kE + k0, 16);
        c[0][jb] = __builtin_amdgcn_wmma_f32_16x16x32_f16(
            false, a, false, b0, (short)0, c[0][jb], false, false);
        c[1][jb] = __builtin_amdgcn_wmma_f32_16x16x32_f16(
            false, a, false, b1, (short)0, c[1][jb], false, false);
      }
    }
    const float Ph0 = P[(size_t)i * kH + col0];
    const float Ph1 = P[(size_t)i * kH + col1];
#pragma unroll
    for (int jb = 0; jb < 7; ++jb) {
#pragma unroll
      for (int v = 0; v < 8; ++v) {
        int j = jb * 16 + hs * 8 + v;
        if (j < kM && ok[j]) {
          float base0 = Ph0 + Q[(size_t)j * kH + col0] + c[0][jb][v];
          float base1 = Ph1 + Q[(size_t)j * kH + col1] + c[1][jb][v];
          const float* lg = logits + ((size_t)i * kM + j) * kT;
#pragma unroll
          for (int t = 0; t < kT; ++t) {
            float lt = lg[t];
            if (lt > 0.f) {
              acc0 += fmaxf(base0 + lt * Wt[t * kH + col0], 0.f);
              acc1 += fmaxf(base1 + lt * Wt[t * kH + col1], 0.f);
            }
          }
        }
      }
    }
  }
  acc0 += __shfl_xor(acc0, 16, 32);   // fold rows 8..15 half onto 0..7 half
  acc1 += __shfl_xor(acc1, 16, 32);
  if (hs == 0) {
    int ae = *any_edge;
    float p0 = child_prev[(size_t)i * kH + col0];
    float p1 = child_prev[(size_t)i * kH + col1];
    child_next[(size_t)i * kH + col0] = ae ? acc0 : p0;
    child_next[(size_t)i * kH + col1] = ae ? acc1 : p1;
  }
}

// ---------------------------------------------------------------------------
// f32 weight [K][N] -> f16 transposed Bt [Npad][K] (zero pad n>=N)
// ---------------------------------------------------------------------------
__global__ void convT_kernel(const float* __restrict__ W, int K, int N,
                             _Float16* __restrict__ Bt, int Npad) {
  int idx = blockIdx.x * blockDim.x + threadIdx.x;
  if (idx >= Npad * K) return;
  int n = idx % Npad, k = idx / Npad;
  float v = (n < N) ? W[(size_t)k * N + n] : 0.f;
  Bt[(size_t)n * K + k] = (_Float16)v;
}

// f32 activation [Mreal][K] -> f16 [Mpad][ldd] at column offset (zero pad rows)
__global__ void convA_kernel(const float* __restrict__ A, int Mreal, int Mpad,
                             int K, _Float16* __restrict__ dst, int ldd,
                             int colOff) {
  int idx = blockIdx.x * blockDim.x + threadIdx.x;
  if (idx >= Mpad * K) return;
  int m = idx / K, k = idx % K;
  float v = (m < Mreal) ? A[(size_t)m * K + k] : 0.f;
  dst[(size_t)m * ldd + colOff + k] = (_Float16)v;
}

__global__ void copy2d_kernel(const float* __restrict__ src, int lds,
                              float* __restrict__ dst, int ldd,
                              int rows, int cols) {
  int idx = blockIdx.x * blockDim.x + threadIdx.x;
  if (idx >= rows * cols) return;
  int r = idx / cols, c = idx % cols;
  dst[(size_t)r * ldd + c] = src[(size_t)r * lds + c];
}

// ---------------------------------------------------------------------------
// Host launcher
// ---------------------------------------------------------------------------
extern "C" void kernel_launch(void* const* d_in, const int* in_sizes, int n_in,
                              void* d_out, int out_size, void* d_ws, size_t ws_size,
                              hipStream_t stream) {
  (void)in_sizes; (void)n_in; (void)out_size; (void)ws_size;
  const float* pf  = (const float*)d_in[0];   // [1,512]
  const float* Wp  = (const float*)d_in[1];   // [512, 51200]
  const float* bp  = (const float*)d_in[2];   // [51200]
  const float* Wex = (const float*)d_in[3];   // [512,1]
  const float* bex = (const float*)d_in[4];   // [1]
  const float* Wel = (const float*)d_in[5];   // [1024,512]
  const float* bel = (const float*)d_in[6];   // [512]
  const float* Wee = (const float*)d_in[7];   // [4,512]
  const float* bee = (const float*)d_in[8];   // [4]
  const float* Wne = (const float*)d_in[9];   // [2,1540,512]
  const float* bne = (const float*)d_in[10];  // [2,512]
  const float* Wch = (const float*)d_in[11];  // [1536,512]
  const float* bch = (const float*)d_in[12];  // [512]
  const float* Wsm = (const float*)d_in[13];  // [512,57]
  const float* bsm = (const float*)d_in[14];  // [57]
  const float* Wc2 = (const float*)d_in[15];  // [512,512]
  const float* bc2 = (const float*)d_in[16];  // [512]

  float* out = (float*)d_out;
  float* out_child = out;                 // [100,512]
  float* out_sem   = out + 51200;         // [100,57]
  float* out_ex    = out + 56900;         // [100,1]
  float* out_edge  = out + 57000;         // [100,100,4]

  // --- workspace bump allocator -------------------------------------------
  char* ws = (char*)d_ws;
  size_t off = 0;
  auto alloc = [&](size_t bytes) -> void* {
    void* p = ws + off;
    off = (off + bytes + 255) & ~(size_t)255;
    return p;
  };
  float* child0  = (float*)alloc((size_t)kM * kH * 4);
  float* child1  = (float*)alloc((size_t)kM * kH * 4);
  float* child2  = (float*)alloc((size_t)kM * kH * 4);
  float* Ael     = (float*)alloc((size_t)kMP * kE * 4);
  float* Bel     = (float*)alloc((size_t)kMP * kE * 4);
  float* Pb      = (float*)alloc((size_t)kMP * kH * 4);
  float* Qb      = (float*)alloc((size_t)kMP * kH * 4);
  float* logits  = (float*)alloc((size_t)kM * kM * kT * 4);
  float* hbuf    = (float*)alloc((size_t)kMP * kH * 4);
  float* semtmp  = (float*)alloc((size_t)kMP * kSEMP * 4);
  float* couttmp = (float*)alloc((size_t)kMP * kH * 4);
  int*   any_edge = (int*)alloc(4);
  unsigned char* ok = (unsigned char*)alloc(kMP);
  _Float16* EL16   = (_Float16*)alloc((size_t)kM * kMP * kE * 2);
  _Float16* C16    = (_Float16*)alloc((size_t)kMP * kH * 2);
  _Float16* hcat16 = (_Float16*)alloc((size_t)kMP * 3 * kH * 2);
  _Float16* h16    = (_Float16*)alloc((size_t)kMP * kH * 2);
  _Float16* WelA_t = (_Float16*)alloc((size_t)kE * kH * 2);
  _Float16* WelB_t = (_Float16*)alloc((size_t)kE * kH * 2);
  _Float16* Wfrom_t = (_Float16*)alloc((size_t)kH * kH * 2);
  _Float16* Wto_t   = (_Float16*)alloc((size_t)kH * kH * 2);
  _Float16* We_t    = (_Float16*)alloc((size_t)kH * kE * 2);
  _Float16* Wch_t   = (_Float16*)alloc((size_t)kH * 3 * kH * 2);
  _Float16* Wsm_t   = (_Float16*)alloc((size_t)kSEMP * kH * 2);
  _Float16* Wc2_t   = (_Float16*)alloc((size_t)kH * kH * 2);

  auto blocks = [](int n) { return (n + 255) / 256; };
  auto convT = [&](const float* W, int K, int N, _Float16* Bt, int Npad) {
    convT_kernel<<<blocks(Npad * K), 256, 0, stream>>>(W, K, N, Bt, Npad);
  };
  auto convA = [&](const float* A, int Mreal, int Mpad, int K, _Float16* dst,
                   int ldd, int colOff) {
    convA_kernel<<<blocks(Mpad * K), 256, 0, stream>>>(A, Mreal, Mpad, K, dst,
                                                       ldd, colOff);
  };
  auto gemm = [&](const _Float16* A, int lda, const _Float16* Bt, int ldb,
                  float* C, int ldc, int Mtiles, int Ntiles, int K,
                  const float* bias, int nbias, int relu) {
    int waves = Mtiles * Ntiles;
    gemm_kernel<<<blocks(waves * 32), 256, 0, stream>>>(
        A, lda, Bt, ldb, C, ldc, Mtiles, Ntiles, K, bias, nbias, relu);
  };

  // --- one-time weight conversions (f32 -> transposed f16) ----------------
  convT(Wel,                 kH, kE, WelA_t, kE);   // top H rows
  convT(Wel + (size_t)kH * kE, kH, kE, WelB_t, kE); // bottom H rows
  convT(Wch, 3 * kH, kH, Wch_t, kH);
  convT(Wsm, kH, kSEM, Wsm_t, kSEMP);
  convT(Wc2, kH, kH, Wc2_t, kH);

  // --- stage 1: children + existence --------------------------------------
  parent_kernel<<<blocks((kM * kH) / 4), 256, 0, stream>>>(pf, Wp, bp, child0);
  exists_kernel<<<blocks(kMP * 32), 256, 0, stream>>>(child0, Wex, bex, out_ex,
                                                      ok, any_edge);

  // --- stage 2: edge latents + edge-exists logits -------------------------
  convA(child0, kM, kMP, kH, C16, kH, 0);
  gemm(C16, kH, WelA_t, kH, Ael, kE, kMP / 16, kE / 16, kH, bel, kE, 0);
  gemm(C16, kH, WelB_t, kH, Bel, kE, kMP / 16, kE / 16, kH, nullptr, 0, 0);
  edge_kernel<<<dim3(kM, kMP), 256, 0, stream>>>(Ael, Bel, Wee, bee, EL16,
                                                 logits, out_edge, ok, any_edge);

  // --- stage 3: message-passing iterations --------------------------------
  for (int it = 0; it < kIT; ++it) {
    const float* Wbase = Wne + (size_t)it * 1540 * kH;
    convT(Wbase,                       kH, kH, Wfrom_t, kH);
    convT(Wbase + (size_t)kH * kH,     kH, kH, Wto_t,   kH);
    convT(Wbase + (size_t)2 * kH * kH, kH, kH, We_t,    kH);
    const float* Wt = Wbase + (size_t)(2 * kH + kE) * kH;  // 4 type rows, f32
    const float* prev = (it == 0) ? child0 : child1;
    float* next = (it == 0) ? child1 : child2;
    convA(prev, kM, kMP, kH, C16, kH, 0);
    gemm(C16, kH, Wfrom_t, kH, Pb, kH, kMP / 16, kH / 16, kH,
         bne + (size_t)it * kH, kH, 0);
    gemm(C16, kH, Wto_t, kH, Qb, kH, kMP / 16, kH / 16, kH, nullptr, 0, 0);
    msg_kernel<<<dim3(kM, kH / 256), 256, 0, stream>>>(
        EL16, We_t, Pb, Qb, Wt, logits, ok, any_edge, prev, next);
  }

  // --- stage 4: head -------------------------------------------------------
  convA(child0, kM, kMP, kH, hcat16, 3 * kH, 0);
  convA(child1, kM, kMP, kH, hcat16, 3 * kH, kH);
  convA(child2, kM, kMP, kH, hcat16, 3 * kH, 2 * kH);
  gemm(hcat16, 3 * kH, Wch_t, 3 * kH, hbuf, kH, kMP / 16, kH / 16, 3 * kH,
       bch, kH, 1);
  convA(hbuf, kMP, kMP, kH, h16, kH, 0);
  gemm(h16, kH, Wsm_t, kH, semtmp, kSEMP, kMP / 16, kSEMP / 16, kH, bsm, kSEM, 0);
  gemm(h16, kH, Wc2_t, kH, couttmp, kH, kMP / 16, kH / 16, kH, bc2, kH, 1);
  copy2d_kernel<<<blocks(kM * kH), 256, 0, stream>>>(couttmp, kH, out_child,
                                                     kH, kM, kH);
  copy2d_kernel<<<blocks(kM * kSEM), 256, 0, stream>>>(semtmp, kSEMP, out_sem,
                                                       kSEM, kM, kSEM);
}
